// Attention_19774029431221
// MI455X (gfx1250) — matrix operand order repaired and checked by reading, not compiled, not run on hardware
//
#include <hip/hip_runtime.h>
#include <hip/hip_bf16.h>

// ---------------------------------------------------------------------------
// Causal flash attention, B=4,H=16,S=2048,D=64, fp32 in/out, gfx1250.
//
// Pre-pass: K -> f16 (row-major), V -> f16 TRANSPOSED [d][s] in d_ws.
// Main:     block = 4 wave32s = 64 query rows; 64-key chunks staged into LDS
//           with global_load_async_to_lds_b128, double-buffered (ASYNCcnt);
//           16 v_wmma_f32_16x16x32_f16 per chunk; sched_barrier keeps the
//           ds_load clauses grouped ahead of the WMMA streams; V fragments
//           prefetched under softmax; masking only in the final chunk;
//           softmax scale/log2e folded into the Q fragment.
// Fallback: self-contained kernel if ws_size is too small for the f16 copies.
// ---------------------------------------------------------------------------

typedef _Float16 v16h __attribute__((ext_vector_type(16)));
typedef _Float16 v8h  __attribute__((ext_vector_type(8)));
typedef _Float16 v4h  __attribute__((ext_vector_type(4)));
typedef float    v8f  __attribute__((ext_vector_type(8)));

#define WMMA_F16F32(a, b, c) \
  __builtin_amdgcn_wmma_f32_16x16x32_f16(false, (a), false, (b), (short)0, (c), false, false)

#define CAT8(lo, hi) __builtin_shufflevector((lo), (hi), 0,1,2,3,4,5,6,7,8,9,10,11,12,13,14,15)

#if __has_builtin(__builtin_amdgcn_sched_barrier)
#define SCHED_FENCE() __builtin_amdgcn_sched_barrier(0)
#else
#define SCHED_FENCE()
#endif

namespace {
constexpr int kB = 4, kH = 16, kS = 2048, kD = 64;
constexpr int BLOCK_M = 64;          // query rows per block (4 waves x 16)
constexpr int CHUNK   = 64;          // keys per iteration (4 score tiles)
constexpr int KSTR    = 72;          // halves per K row in LDS  (144B, 36-bank step)
constexpr int VSTR    = 72;          // halves per V^T row in LDS
constexpr int PSTR    = 72;          // halves per P row in LDS
constexpr float SCALE_LOG2E = 0.125f * 1.44269504088896340736f; // (1/sqrt(64))*log2(e)
}

// ---- gfx1250 async global->LDS copy (ASYNCcnt-tracked) ---------------------
__device__ __forceinline__ void async_b128(unsigned lds_off, const void* gaddr) {
  asm volatile("global_load_async_to_lds_b128 %0, %1, off"
               :: "v"(lds_off), "v"((unsigned long long)gaddr)
               : "memory");
}
__device__ __forceinline__ void wait_async0() {
  asm volatile("s_wait_asynccnt 0x0" ::: "memory");
}
__device__ __forceinline__ unsigned lds_off(const void* p) {
  return (unsigned)(unsigned long long)p;   // low 32 bits of generic LDS ptr
}

// ===========================================================================
// Pre-pass: per (bh, 64-key tile): Kh[s][d] = (f16)K, VTh[d][s] = (f16)V
// ===========================================================================
__global__ __launch_bounds__(128)
void Attention_prepass_kernel(const float* __restrict__ K,
                              const float* __restrict__ V,
                              _Float16* __restrict__ Kh,
                              _Float16* __restrict__ VTh) {
  __shared__ __align__(16) _Float16 tileT[64 * 72];   // [d][key], padded
  const int tid  = threadIdx.x;
  const int tile = blockIdx.x % (kS / 64);
  const int bh   = blockIdx.x / (kS / 64);
  const int t0   = tile * 64;

  const float* __restrict__ k = K + (size_t)bh * kS * kD;
  const float* __restrict__ v = V + (size_t)bh * kS * kD;
  _Float16* __restrict__ kh = Kh  + (size_t)bh * kS * kD;
  _Float16* __restrict__ vt = VTh + (size_t)bh * kS * kD;

#pragma unroll
  for (int it = 0; it < 8; ++it) {
    const int idx = tid + it * 128;        // 0..1023 over 64x16 float4s
    const int key = idx >> 4;
    const int d4  = (idx & 15) * 4;
    const float4 kv = *(const float4*)(k + (size_t)(t0 + key) * kD + d4);
    *(v4h*)(kh + (size_t)(t0 + key) * kD + d4) =
        (v4h){(_Float16)kv.x, (_Float16)kv.y, (_Float16)kv.z, (_Float16)kv.w};
    const float4 vv = *(const float4*)(v + (size_t)(t0 + key) * kD + d4);
    tileT[(d4 + 0) * 72 + key] = (_Float16)vv.x;
    tileT[(d4 + 1) * 72 + key] = (_Float16)vv.y;
    tileT[(d4 + 2) * 72 + key] = (_Float16)vv.z;
    tileT[(d4 + 3) * 72 + key] = (_Float16)vv.w;
  }
  __syncthreads();
#pragma unroll
  for (int it = 0; it < 4; ++it) {
    const int idx = tid + it * 128;        // 0..511: 64 d-rows x 8 v8h segs
    const int d   = idx >> 3;
    const int seg = idx & 7;
    *(v8h*)(vt + (size_t)d * kS + t0 + seg * 8) = *(const v8h*)&tileT[d * 72 + seg * 8];
  }
}

// ===========================================================================
// Shared compute body: one 64-key chunk for one 16-row wave tile
// ===========================================================================
template <bool MASKED>
__device__ __forceinline__ void fa_body(const _Float16* __restrict__ sKb,
                                        const _Float16* __restrict__ sVTb,
                                        _Float16* __restrict__ pb,
                                        const v16h aq[2], v8f acc[4],
                                        float mrow[8], float lrow[8],
                                        int kb, int q0, int wave, int nloc, bool hiHalf) {
  const int kb2 = hiHalf ? 8 : 0;

  // ---- batch-load ALL K B-fragments (one ds_load clause) ------------------
  v16h bk[4][2];
#pragma unroll
  for (int t = 0; t < 4; ++t)
#pragma unroll
    for (int c = 0; c < 2; ++c) {
      const int key = t * 16 + nloc;
      const int db  = c * 32 + kb2;
      const v8h lo = *(const v8h*)&sKb[key * KSTR + db];
      const v8h hi = *(const v8h*)&sKb[key * KSTR + db + 16];
      bk[t][c] = CAT8(lo, hi);
    }
  SCHED_FENCE();   // keep all loads above; WMMAs stream with partial waits

  // ---- 8 QK WMMAs back-to-back --------------------------------------------
  v8f ct[4];
#pragma unroll
  for (int t = 0; t < 4; ++t) {
    v8f cacc = {};
    cacc = WMMA_F16F32(aq[0], bk[t][0], cacc);
    cacc = WMMA_F16F32(aq[1], bk[t][1], cacc);
    ct[t] = cacc;
  }
  SCHED_FENCE();

  // ---- prefetch ALL V B-fragments; loads fly under the softmax VALU --------
  v16h bv[4][2];
#pragma unroll
  for (int g = 0; g < 4; ++g) {
    const int f = g * 16 + nloc;
#pragma unroll
    for (int h = 0; h < 2; ++h) {
      const v8h lo = *(const v8h*)&sVTb[f * VSTR + h * 32 + kb2];
      const v8h hi = *(const v8h*)&sVTb[f * VSTR + h * 32 + kb2 + 16];
      bv[g][h] = CAT8(lo, hi);
    }
  }
  SCHED_FENCE();   // V loads issued here; softmax below hides their latency

  // ---- masked exp2-domain online softmax ----------------------------------
  float s[4][8], hm[8];
#pragma unroll
  for (int r = 0; r < 8; ++r) {
    const int qr = q0 + wave * 16 + (hiHalf ? 8 : 0) + r;
#pragma unroll
    for (int t = 0; t < 4; ++t) {
      if (MASKED) {
        const int kc = kb + t * 16 + nloc;
        s[t][r] = (kc <= qr) ? ct[t][r] : -__builtin_inff();
      } else {
        s[t][r] = ct[t][r];
      }
    }
    hm[r] = fmaxf(fmaxf(s[0][r], s[1][r]), fmaxf(s[2][r], s[3][r]));
  }
#pragma unroll
  for (int off = 8; off >= 1; off >>= 1)      // stays inside 16-lane half
#pragma unroll
    for (int r = 0; r < 8; ++r) hm[r] = fmaxf(hm[r], __shfl_xor(hm[r], off, 32));

  float alpha[8], rs[8];
#pragma unroll
  for (int r = 0; r < 8; ++r) {
    const float mn = fmaxf(mrow[r], hm[r]);
    alpha[r] = __builtin_amdgcn_exp2f(mrow[r] - mn);
    mrow[r]  = mn;
#pragma unroll
    for (int t = 0; t < 4; ++t) s[t][r] = __builtin_amdgcn_exp2f(s[t][r] - mn);
    rs[r] = (s[0][r] + s[1][r]) + (s[2][r] + s[3][r]);
  }
#pragma unroll
  for (int off = 8; off >= 1; off >>= 1)
#pragma unroll
    for (int r = 0; r < 8; ++r) rs[r] += __shfl_xor(rs[r], off, 32);
#pragma unroll
  for (int r = 0; r < 8; ++r) lrow[r] = lrow[r] * alpha[r] + rs[r];
#pragma unroll
  for (int g = 0; g < 4; ++g)
#pragma unroll
    for (int r = 0; r < 8; ++r) acc[g][r] *= alpha[r];

  // ---- P: C-layout -> A-layout via per-wave LDS (same-wave DS is in-order)
#pragma unroll
  for (int r = 0; r < 8; ++r) {
    const int rr = r + (hiHalf ? 8 : 0);
#pragma unroll
    for (int t = 0; t < 4; ++t)
      pb[rr * PSTR + t * 16 + nloc] = (_Float16)s[t][r];
  }
  const int col0 = hiHalf ? 8 : 0;
  v16h ap[2];
#pragma unroll
  for (int h = 0; h < 2; ++h) {
    const v8h plo = *(const v8h*)&pb[nloc * PSTR + h * 32 + col0];
    const v8h phi = *(const v8h*)&pb[nloc * PSTR + h * 32 + col0 + 16];
    ap[h] = CAT8(plo, phi);
  }
  SCHED_FENCE();

  // ---- O += P V : 8 WMMAs with fragments already in registers -------------
#pragma unroll
  for (int g = 0; g < 4; ++g) {
    acc[g] = WMMA_F16F32(ap[0], bv[g][0], acc[g]);
    acc[g] = WMMA_F16F32(ap[1], bv[g][1], acc[g]);
  }
}

__device__ __forceinline__ void load_q_frags(const float* __restrict__ q,
                                             int qrow_lane, bool hiHalf, v16h aq[2]) {
#pragma unroll
  for (int c = 0; c < 2; ++c) {
    const int d0 = c * 32 + (hiHalf ? 8 : 0);
    const float4 f0 = *(const float4*)(q + (size_t)qrow_lane * kD + d0);
    const float4 f1 = *(const float4*)(q + (size_t)qrow_lane * kD + d0 + 4);
    const float4 f2 = *(const float4*)(q + (size_t)qrow_lane * kD + d0 + 16);
    const float4 f3 = *(const float4*)(q + (size_t)qrow_lane * kD + d0 + 20);
    const float sc = SCALE_LOG2E;
    aq[c] = (v16h){(_Float16)(f0.x*sc), (_Float16)(f0.y*sc), (_Float16)(f0.z*sc), (_Float16)(f0.w*sc),
                   (_Float16)(f1.x*sc), (_Float16)(f1.y*sc), (_Float16)(f1.z*sc), (_Float16)(f1.w*sc),
                   (_Float16)(f2.x*sc), (_Float16)(f2.y*sc), (_Float16)(f2.z*sc), (_Float16)(f2.w*sc),
                   (_Float16)(f3.x*sc), (_Float16)(f3.y*sc), (_Float16)(f3.z*sc), (_Float16)(f3.w*sc)};
  }
}

// ===========================================================================
// Main kernel (async staging, double-buffered, masked chunk peeled)
// ===========================================================================
__device__ __forceinline__ void stage_chunk_async(const _Float16* kh, const _Float16* vt,
                                                  int kb, _Float16* sKb, _Float16* sVTb,
                                                  int tid) {
  // K chunk: 64 rows x 128B = 512 b128 transfers
#pragma unroll
  for (int it = 0; it < 4; ++it) {
    const int idx = tid + it * 128;
    const int row = idx >> 3, seg = idx & 7;
    async_b128(lds_off(&sKb[row * KSTR + seg * 8]),
               (const char*)kh + (size_t)(kb + row) * 128 + seg * 16);
  }
  // V^T chunk: 64 feature rows x 128B = 512 b128 transfers
#pragma unroll
  for (int it = 0; it < 4; ++it) {
    const int idx = tid + it * 128;
    const int f = idx >> 3, seg = idx & 7;
    async_b128(lds_off(&sVTb[f * VSTR + seg * 8]),
               (const char*)vt + (size_t)f * (kS * 2) + (size_t)kb * 2 + seg * 16);
  }
}

__global__ __launch_bounds__(128)
void Attention_19774029431221_kernel(const float* __restrict__ Q,
                                     const _Float16* __restrict__ Kh,
                                     const _Float16* __restrict__ VTh,
                                     float* __restrict__ O) {
  __shared__ __align__(16) _Float16 sK [2][CHUNK * KSTR];
  __shared__ __align__(16) _Float16 sVT[2][kD    * VSTR];
  __shared__ __align__(16) _Float16 sP [4 * 16 * PSTR];

  const int tid  = threadIdx.x;
  const int lane = tid & 31;
  const int wave = tid >> 5;
  const int nloc = lane & 15;
  const bool hiHalf = lane >= 16;

  const int qt = blockIdx.x % (kS / BLOCK_M);
  const int bh = blockIdx.x / (kS / BLOCK_M);
  const int q0 = qt * BLOCK_M;

  const float* __restrict__ q = Q + (size_t)bh * kS * kD;
  const _Float16* __restrict__ kh = Kh  + (size_t)bh * kS * kD;
  const _Float16* __restrict__ vt = VTh + (size_t)bh * kS * kD;
  float* __restrict__ o = O + (size_t)bh * kS * kD;

  v16h aq[2];
  load_q_frags(q, q0 + wave * 16 + nloc, hiHalf, aq);

  float mrow[8], lrow[8];
  v8f acc[4];
#pragma unroll
  for (int r = 0; r < 8; ++r) { mrow[r] = -__builtin_inff(); lrow[r] = 0.0f; }
#pragma unroll
  for (int g = 0; g < 4; ++g)
#pragma unroll
    for (int r = 0; r < 8; ++r) acc[g][r] = 0.0f;

  _Float16* pb = &sP[wave * 16 * PSTR];
  const int ncb = qt + 1;                    // 64-key chunks up to diagonal

  stage_chunk_async(kh, vt, 0, sK[0], sVT[0], tid);   // prefetch chunk 0

  for (int cb = 0; cb < ncb; ++cb) {
    wait_async0();        // this wave's fills for chunk cb have landed
    __syncthreads();      // all waves: fills done AND chunk cb-1 reads done
    if (cb + 1 < ncb)     // overlap next fill with this chunk's math
      stage_chunk_async(kh, vt, (cb + 1) * CHUNK, sK[(cb + 1) & 1], sVT[(cb + 1) & 1], tid);

    const int kb = cb * CHUNK;
    if (cb < ncb - 1)
      fa_body<false>(sK[cb & 1], sVT[cb & 1], pb, aq, acc, mrow, lrow,
                     kb, q0, wave, nloc, hiHalf);
    else
      fa_body<true >(sK[cb & 1], sVT[cb & 1], pb, aq, acc, mrow, lrow,
                     kb, q0, wave, nloc, hiHalf);
  }

#pragma unroll
  for (int g = 0; g < 4; ++g)
#pragma unroll
    for (int r = 0; r < 8; ++r) {
      const int qr = q0 + wave * 16 + (hiHalf ? 8 : 0) + r;
      o[(size_t)qr * kD + g * 16 + nloc] = acc[g][r] / lrow[r];
    }
}

// ===========================================================================
// Fallback (self-contained, no workspace)
// ===========================================================================
__global__ __launch_bounds__(128)
void Attention_fallback_kernel(const float* __restrict__ Q,
                               const float* __restrict__ K,
                               const float* __restrict__ V,
                               float* __restrict__ O) {
  __shared__ __align__(16) _Float16 sK [CHUNK * KSTR];
  __shared__ __align__(16) _Float16 sVT[kD    * VSTR];
  __shared__ __align__(16) _Float16 sP [4 * 16 * PSTR];

  const int tid  = threadIdx.x;
  const int lane = tid & 31;
  const int wave = tid >> 5;
  const int nloc = lane & 15;
  const bool hiHalf = lane >= 16;

  const int qt = blockIdx.x % (kS / BLOCK_M);
  const int bh = blockIdx.x / (kS / BLOCK_M);
  const int q0 = qt * BLOCK_M;

  const float* __restrict__ q = Q + (size_t)bh * kS * kD;
  const float* __restrict__ k = K + (size_t)bh * kS * kD;
  const float* __restrict__ v = V + (size_t)bh * kS * kD;
  float* __restrict__ o       = O + (size_t)bh * kS * kD;

  v16h aq[2];
  load_q_frags(q, q0 + wave * 16 + nloc, hiHalf, aq);

  float mrow[8], lrow[8];
  v8f acc[4];
#pragma unroll
  for (int r = 0; r < 8; ++r) { mrow[r] = -__builtin_inff(); lrow[r] = 0.0f; }
#pragma unroll
  for (int g = 0; g < 4; ++g)
#pragma unroll
    for (int r = 0; r < 8; ++r) acc[g][r] = 0.0f;

  _Float16* pb = &sP[wave * 16 * PSTR];
  const int ncb = qt + 1;

  for (int cb = 0; cb < ncb; ++cb) {
    const int kb = cb * CHUNK;
    __syncthreads();
#pragma unroll
    for (int it = 0; it < 8; ++it) {
      const int idx = tid + it * 128;       // 0..1023 over 64x16 float4s
      const int kl  = idx >> 4;
      const int d4  = (idx & 15) * 4;
      const float4 kv = *(const float4*)(k + (size_t)(kb + kl) * kD + d4);
      *(v4h*)&sK[kl * KSTR + d4] =
          (v4h){(_Float16)kv.x, (_Float16)kv.y, (_Float16)kv.z, (_Float16)kv.w};
      const float4 vv = *(const float4*)(v + (size_t)(kb + kl) * kD + d4);
      sVT[(d4 + 0) * VSTR + kl] = (_Float16)vv.x;
      sVT[(d4 + 1) * VSTR + kl] = (_Float16)vv.y;
      sVT[(d4 + 2) * VSTR + kl] = (_Float16)vv.z;
      sVT[(d4 + 3) * VSTR + kl] = (_Float16)vv.w;
    }
    __syncthreads();

    if (cb < ncb - 1)
      fa_body<false>(sK, sVT, pb, aq, acc, mrow, lrow, kb, q0, wave, nloc, hiHalf);
    else
      fa_body<true >(sK, sVT, pb, aq, acc, mrow, lrow, kb, q0, wave, nloc, hiHalf);
  }

#pragma unroll
  for (int g = 0; g < 4; ++g)
#pragma unroll
    for (int r = 0; r < 8; ++r) {
      const int qr = q0 + wave * 16 + (hiHalf ? 8 : 0) + r;
      o[(size_t)qr * kD + g * 16 + nloc] = acc[g][r] / lrow[r];
    }
}

// ===========================================================================
extern "C" void kernel_launch(void* const* d_in, const int* in_sizes, int n_in,
                              void* d_out, int out_size, void* d_ws, size_t ws_size,
                              hipStream_t stream) {
  (void)in_sizes; (void)n_in; (void)out_size;
  const float* q = (const float*)d_in[0];
  const float* k = (const float*)d_in[1];
  const float* v = (const float*)d_in[2];
  // d_in[3] is the causal mask; causality is computed analytically in-kernel.
  float* out = (float*)d_out;

  const int grid = kB * kH * (kS / BLOCK_M);                 // 2048 blocks
  const size_t elems = (size_t)kB * kH * kS * kD;
  const size_t need  = 2 * elems * sizeof(_Float16);         // Kh + VTh

  if (d_ws != nullptr && ws_size >= need) {
    _Float16* Kh  = (_Float16*)d_ws;
    _Float16* VTh = Kh + elems;
    Attention_prepass_kernel<<<grid, 128, 0, stream>>>(k, v, Kh, VTh);
    Attention_19774029431221_kernel<<<grid, 128, 0, stream>>>(q, Kh, VTh, out);
  } else {
    Attention_fallback_kernel<<<grid, 128, 0, stream>>>(q, k, v, out);
  }
}
